// LmLSTMSharedEmb_62775241999108
// MI455X (gfx1250) — matrix operand order repaired and checked by reading, not compiled, not run on hardware
//
#include <hip/hip_runtime.h>

// ---------------------------------------------------------------------------
// Types for CDNA5 WMMA (wave32, 16x16x32 bf16 -> f32)
// ---------------------------------------------------------------------------
typedef __bf16 bf16;
typedef __attribute__((ext_vector_type(16))) __bf16 v16bf;
typedef __attribute__((ext_vector_type(8)))  float  v8f;
typedef __attribute__((ext_vector_type(8)))  unsigned short u16x8; // 16B chunk

union Frag {
    v16bf v;
    u16x8 q[2];
};

__device__ __forceinline__ v8f v8f_zero() {
    v8f z;
#pragma unroll
    for (int i = 0; i < 8; ++i) z[i] = 0.0f;
    return z;
}

__device__ __forceinline__ float sigmoidf_dev(float x) {
    return 1.0f / (1.0f + __expf(-x));
}

// ---------------------------------------------------------------------------
// Model dims
// ---------------------------------------------------------------------------
#define BB    4
#define SS    512
#define EMBD  512
#define HID   1024
#define G4    4096        // 4*HID
#define VOCAB 32000
#define MROWS 2048        // B*S

// ---------------------------------------------------------------------------
// Fragment loaders.
// A (16-bit 16x32 layout): lane (half,lm) holds row lm; elems 0..7 -> K=k0+8*half+e,
//   elems 8..15 -> K=k0+8*half+16+(e-8): two 16B loads at +k0 and +k0+16.
// B (32x16 layout) from [N,K]-major storage: lane holds col lm; same K order as A:
//   two 16B loads at +k0 and +k0+16 from the weight row.
// ---------------------------------------------------------------------------
__device__ __forceinline__ void load_a2(Frag* a, const bf16* aRow0, const bf16* aRow1, int k0) {
    a[0].q[0] = *(const u16x8*)(aRow0 + k0);
    a[0].q[1] = *(const u16x8*)(aRow0 + k0 + 16);
    a[1].q[0] = *(const u16x8*)(aRow1 + k0);
    a[1].q[1] = *(const u16x8*)(aRow1 + k0 + 16);
}

__device__ __forceinline__ void load_b4(Frag* b, const bf16* const* bRow, int k0) {
#pragma unroll
    for (int t = 0; t < 4; ++t) {
        b[t].q[0] = *(const u16x8*)(bRow[t] + k0);
        b[t].q[1] = *(const u16x8*)(bRow[t] + k0 + 16);
    }
}

__device__ __forceinline__ void mma_2x4(v8f acc[2][4], const Frag* a, const Frag* b) {
#pragma unroll
    for (int mi = 0; mi < 2; ++mi)
#pragma unroll
        for (int ni = 0; ni < 4; ++ni)
            acc[mi][ni] = __builtin_amdgcn_wmma_f32_16x16x32_bf16(
                false, a[mi].v, false, b[ni].v, (short)0, acc[mi][ni], false, false);
}

// ---------------------------------------------------------------------------
// Utility kernels: convert / transpose-convert / embed / zero
// ---------------------------------------------------------------------------
__global__ void convert_f32_bf16(const float* __restrict__ in, bf16* __restrict__ out, long n) {
    long i = (long)blockIdx.x * blockDim.x + threadIdx.x;
    if (i < n) out[i] = (bf16)in[i];
}

// in: f32 [K,N] row-major; out: bf16 [N,K] row-major (i.e. B stored N-major)
__global__ void transpose_convert(const float* __restrict__ in, bf16* __restrict__ out,
                                  int K, int N) {
    long i = (long)blockIdx.x * blockDim.x + threadIdx.x;
    long total = (long)K * N;
    if (i >= total) return;
    int k = (int)(i % K);       // consecutive threads -> consecutive k -> coalesced writes
    int n = (int)(i / K);
    out[(long)n * K + k] = (bf16)in[(long)k * N + n];
}

// x_bf16[row, :] = bf16(emb[tok[row], :]) , row in [0, 2048)
__global__ void embed_kernel(const int* __restrict__ tok, const float* __restrict__ emb,
                             bf16* __restrict__ xbf) {
    int row = blockIdx.x;
    int t = tok[row];
    const float* src = emb + (long)t * EMBD;
    bf16* dst = xbf + (long)row * EMBD;
    for (int c = threadIdx.x; c < EMBD; c += blockDim.x) dst[c] = (bf16)src[c];
}

__global__ void zero_f32(float* __restrict__ p, int n) {
    int i = blockIdx.x * blockDim.x + threadIdx.x;
    if (i < n) p[i] = 0.0f;
}

// ---------------------------------------------------------------------------
// Generic bf16 WMMA GEMM:  C[M,N] = A[M,K] * B[K,N] (+bias)
//   A: bf16 row-major [M,K], lda = K
//   Bt: bf16 [N,K] row-major (B transposed), ldb = K
// Wave tile: 32M x 64N (2 A-frags x 4 B-frags -> 8 WMMA per k-step).
// Block = 128 threads (4 waves, stacked in M) => 128M x 64N tile.
// Grid = (N/64, M/128). K must be a multiple of 64 (two-stage pipelined loop).
// Software pipelining: next k-step's fragments are loaded before the current
// step's WMMAs so loads overlap the MMA group; B stream prefetched ahead.
// MODE 0: f32 out (+bias)   -> Cf
// MODE 1: bf16 out (+bias)  -> Cb
// MODE 2: f32 out, logits mask epilogue (PAD rows -> onehot0) -> Cf
// ---------------------------------------------------------------------------
template<int MODE>
__global__ void __launch_bounds__(128)
gemm_bt(const bf16* __restrict__ A, int lda,
        const bf16* __restrict__ Bt, int ldb,
        int M, int N, int K,
        const float* __restrict__ bias,
        float* __restrict__ Cf,
        bf16* __restrict__ Cb,
        const int* __restrict__ maskTok) {
    const int lane = threadIdx.x & 31;
    const int wave = threadIdx.x >> 5;
    const int half = lane >> 4;       // 0: lanes 0-15, 1: lanes 16-31
    const int lm   = lane & 15;       // A row within tile / B,C column within tile
    const int n0   = blockIdx.x * 64;
    const int m0   = blockIdx.y * 128 + wave * 32;

    v8f acc[2][4];
#pragma unroll
    for (int mi = 0; mi < 2; ++mi)
#pragma unroll
        for (int ni = 0; ni < 4; ++ni) acc[mi][ni] = v8f_zero();

    const bf16* aRow0 = A + (long)(m0 + lm) * lda + 8 * half;
    const bf16* aRow1 = aRow0 + (long)16 * lda;
    const bf16* bRow[4];
#pragma unroll
    for (int t = 0; t < 4; ++t)
        bRow[t] = Bt + (long)(n0 + t * 16 + lm) * ldb + 8 * half;

    Frag a0[2], a1[2], b0[4], b1[4];
    load_a2(a0, aRow0, aRow1, 0);
    load_b4(b0, bRow, 0);

    for (int k0 = 0; k0 < K; k0 += 64) {
        // stage 0: issue loads for k0+32, prefetch far B stream, compute k0
        load_a2(a1, aRow0, aRow1, k0 + 32);
        load_b4(b1, bRow, k0 + 32);
        if (k0 + 128 < K) {
#pragma unroll
            for (int t = 0; t < 4; ++t) __builtin_prefetch(bRow[t] + k0 + 128, 0, 1);
        }
        mma_2x4(acc, a0, b0);
        // stage 1: issue loads for k0+64, compute k0+32
        if (k0 + 64 < K) {
            load_a2(a0, aRow0, aRow1, k0 + 64);
            load_b4(b0, bRow, k0 + 64);
        }
        mma_2x4(acc, a1, b1);
    }

    // C/D layout: VGPR r of tile (mi,t) holds row m0 + mi*16 + r + 8*half, col n0+t*16+lm
#pragma unroll
    for (int mi = 0; mi < 2; ++mi) {
#pragma unroll
        for (int t = 0; t < 4; ++t) {
            const int n = n0 + t * 16 + lm;
            const float bv = (MODE == 2) ? 0.0f : (bias ? bias[n] : 0.0f);
#pragma unroll
            for (int r = 0; r < 8; ++r) {
                const int m = m0 + mi * 16 + r + 8 * half;
                float v = acc[mi][t][r] + bv;
                if (MODE == 0) {
                    Cf[(long)m * N + n] = v;
                } else if (MODE == 1) {
                    Cb[(long)m * N + n] = (bf16)v;
                } else {
                    const bool valid = maskTok[m] != 0;
                    Cf[(long)m * N + n] = valid ? v : (n == 0 ? 1.0f : 0.0f);
                }
            }
        }
    }
}

// ---------------------------------------------------------------------------
// One LSTM timestep. Grid: 64 blocks x 32 threads (1 wave). Block j handles
// hidden units [j*16, j*16+16). Computes z = h_{t-1} @ Wr (4 gate tiles via
// WMMA, batch rows 0..3 padded to 16 with zeros), adds precomputed
// Xk = x@Wk + b, applies gates, updates c (f32 ping-pong) and writes h (bf16)
// into the layer output. Double-buffered k-loop (HID = 1024 = 16 x 64).
//   hprev: bf16, rows strided by hstride (== y + (t-1)*HID); hvalid=0 => h==0
//   Wrt:   bf16 [4096, 1024]  (Wr transposed, gate-col major)
//   yout:  bf16, == y + t*HID, rows strided by hstride
// ---------------------------------------------------------------------------
__global__ void __launch_bounds__(32)
lstm_step(const bf16* __restrict__ hprev, int hvalid, long hstride,
          const float* __restrict__ cin, float* __restrict__ cout,
          const float* __restrict__ xk,     // [2048, 4096], already includes bias
          const bf16* __restrict__ Wrt,
          bf16* __restrict__ yout,
          const int* __restrict__ tok, int t) {
    const int lane = threadIdx.x;
    const int half = lane >> 4;
    const int lm   = lane & 15;
    const int j0   = blockIdx.x * 16;

    v8f acc[4];
#pragma unroll
    for (int g = 0; g < 4; ++g) acc[g] = v8f_zero();

    if (hvalid) {
        const bool arow_valid = (lm < 4);            // batch rows 0..3, rest zero-pad
        const bf16* aRow = hprev + (long)lm * hstride + 8 * half;
        const bf16* bRow[4];
#pragma unroll
        for (int g = 0; g < 4; ++g)
            bRow[g] = Wrt + (long)(g * HID + j0 + lm) * HID + 8 * half;

        Frag a0, a1, b0[4], b1[4];
#pragma unroll
        for (int e = 0; e < 16; ++e) { a0.v[e] = (bf16)0.0f; a1.v[e] = (bf16)0.0f; }
        if (arow_valid) {
            a0.q[0] = *(const u16x8*)(aRow);
            a0.q[1] = *(const u16x8*)(aRow + 16);
        }
        load_b4(b0, bRow, 0);

        for (int k0 = 0; k0 < HID; k0 += 64) {
            // stage 0: loads for k0+32 (always < HID), prefetch, compute k0
            if (arow_valid) {
                a1.q[0] = *(const u16x8*)(aRow + k0 + 32);
                a1.q[1] = *(const u16x8*)(aRow + k0 + 48);
            }
            load_b4(b1, bRow, k0 + 32);
            if (k0 + 128 < HID) {
#pragma unroll
                for (int g = 0; g < 4; ++g) __builtin_prefetch(bRow[g] + k0 + 128, 0, 1);
            }
#pragma unroll
            for (int g = 0; g < 4; ++g)
                acc[g] = __builtin_amdgcn_wmma_f32_16x16x32_bf16(
                    false, a0.v, false, b0[g].v, (short)0, acc[g], false, false);
            // stage 1: loads for k0+64, compute k0+32
            if (k0 + 64 < HID) {
                if (arow_valid) {
                    a0.q[0] = *(const u16x8*)(aRow + k0 + 64);
                    a0.q[1] = *(const u16x8*)(aRow + k0 + 80);
                }
                load_b4(b0, bRow, k0 + 64);
            }
#pragma unroll
            for (int g = 0; g < 4; ++g)
                acc[g] = __builtin_amdgcn_wmma_f32_16x16x32_bf16(
                    false, a1.v, false, b1[g].v, (short)0, acc[g], false, false);
        }
    }

    // Batch rows 0..3 live in acc[g][0..3] on lanes with half==0 (column = lm).
    if (half == 0) {
        const int j = j0 + lm;
#pragma unroll
        for (int b = 0; b < 4; ++b) {
            const long zrow = ((long)b * SS + t) * G4;
            float zi = acc[0][b] + xk[zrow + 0 * HID + j];
            float zf = acc[1][b] + xk[zrow + 1 * HID + j];
            float zg = acc[2][b] + xk[zrow + 2 * HID + j];
            float zo = acc[3][b] + xk[zrow + 3 * HID + j];
            float ig = sigmoidf_dev(zi);
            float fg = sigmoidf_dev(zf);
            float gg = tanhf(zg);
            float og = sigmoidf_dev(zo);
            float cp = cin[b * HID + j];
            float cn = fg * cp + ig * gg;
            float hn = og * tanhf(cn);
            bool  m  = tok[b * SS + t] != 0;
            float hp = hvalid ? (float)hprev[(long)b * hstride + j] : 0.0f;
            float h  = m ? hn : hp;
            cout[b * HID + j] = m ? cn : cp;
            yout[(long)b * hstride + j] = (bf16)h;
        }
    }
}

// ---------------------------------------------------------------------------
// Host orchestration
// ---------------------------------------------------------------------------
extern "C" void kernel_launch(void* const* d_in, const int* in_sizes, int n_in,
                              void* d_out, int out_size, void* d_ws, size_t ws_size,
                              hipStream_t stream) {
    (void)in_sizes; (void)n_in; (void)out_size; (void)ws_size;

    const int*   inputs = (const int*)  d_in[0];
    const float* emb    = (const float*)d_in[1];
    const float* Wk0    = (const float*)d_in[2];
    const float* Wr0    = (const float*)d_in[3];
    const float* b0     = (const float*)d_in[4];
    const float* Wk1    = (const float*)d_in[5];
    const float* Wr1    = (const float*)d_in[6];
    const float* b1     = (const float*)d_in[7];
    const float* Wp     = (const float*)d_in[8];
    const float* bp     = (const float*)d_in[9];
    float* out = (float*)d_out;

    // Workspace carve-up (all sizes 256B-aligned)
    char* ws = (char*)d_ws;
    size_t off = 0;
    auto carve = [&](size_t bytes) -> char* {
        char* p = ws + off;
        off += (bytes + 255) & ~(size_t)255;
        return p;
    };
    bf16*  xbf   = (bf16*) carve((size_t)MROWS * EMBD * 2);   //  2 MB  embedded inputs
    bf16*  embbf = (bf16*) carve((size_t)VOCAB * EMBD * 2);   // 33 MB  tied-emb B matrix [N,K]
    bf16*  Wk0t  = (bf16*) carve((size_t)G4 * EMBD * 2);      //  4 MB
    bf16*  Wr0t  = (bf16*) carve((size_t)G4 * HID * 2);       //  8 MB
    bf16*  Wk1t  = (bf16*) carve((size_t)G4 * HID * 2);       //  8 MB
    bf16*  Wr1t  = (bf16*) carve((size_t)G4 * HID * 2);       //  8 MB
    bf16*  Wpt   = (bf16*) carve((size_t)EMBD * HID * 2);     //  1 MB
    float* Xk    = (float*)carve((size_t)MROWS * G4 * 4);     // 34 MB  (reused layer0/layer1)
    bf16*  y0    = (bf16*) carve((size_t)MROWS * HID * 2);    //  4 MB
    bf16*  y1    = (bf16*) carve((size_t)MROWS * HID * 2);    //  4 MB
    bf16*  P     = (bf16*) carve((size_t)MROWS * EMBD * 2);   //  2 MB
    float* cbuf  = (float*)carve((size_t)2 * BB * HID * 4);   // ping-pong cell state

    const long hstride = (long)SS * HID;  // row stride of y buffers (per batch)

    // --- Stage 0: weight conversion (one-time per call) ---
    {
        long n = (long)VOCAB * EMBD;
        convert_f32_bf16<<<dim3((unsigned)((n + 255) / 256)), 256, 0, stream>>>(emb, embbf, n);
    }
    auto tconv = [&](const float* in, bf16* outp, int K, int N) {
        long n = (long)K * N;
        transpose_convert<<<dim3((unsigned)((n + 255) / 256)), 256, 0, stream>>>(in, outp, K, N);
    };
    tconv(Wk0, Wk0t, EMBD, G4);
    tconv(Wr0, Wr0t, HID,  G4);
    tconv(Wk1, Wk1t, HID,  G4);
    tconv(Wr1, Wr1t, HID,  G4);
    tconv(Wp,  Wpt,  HID,  EMBD);

    // --- Stage 1: embedding gather (f32 -> bf16 rows) ---
    embed_kernel<<<dim3(MROWS), 256, 0, stream>>>(inputs, emb, xbf);

    // --- Stage 2: layer 0 input GEMM  Xk = x @ Wk0 + b0  [2048,4096] f32 ---
    gemm_bt<0><<<dim3(G4 / 64, MROWS / 128), 128, 0, stream>>>(
        xbf, EMBD, Wk0t, EMBD, MROWS, G4, EMBD, b0, Xk, nullptr, nullptr);

    // --- Stage 3: layer 0 recurrence (512 dependent steps) ---
    zero_f32<<<dim3((2 * BB * HID + 255) / 256), 256, 0, stream>>>(cbuf, 2 * BB * HID);
    for (int t = 0; t < SS; ++t) {
        const bf16* hp = (t > 0) ? (y0 + (long)(t - 1) * HID) : y0;
        lstm_step<<<dim3(HID / 16), 32, 0, stream>>>(
            hp, t > 0, hstride,
            cbuf + (t & 1) * (BB * HID), cbuf + ((t + 1) & 1) * (BB * HID),
            Xk, Wr0t, y0 + (long)t * HID, inputs, t);
    }

    // --- Stage 4: layer 1 input GEMM  Xk = y0 @ Wk1 + b1 ---
    gemm_bt<0><<<dim3(G4 / 64, MROWS / 128), 128, 0, stream>>>(
        y0, HID, Wk1t, HID, MROWS, G4, HID, b1, Xk, nullptr, nullptr);

    // --- Stage 5: layer 1 recurrence ---
    zero_f32<<<dim3((2 * BB * HID + 255) / 256), 256, 0, stream>>>(cbuf, 2 * BB * HID);
    for (int t = 0; t < SS; ++t) {
        const bf16* hp = (t > 0) ? (y1 + (long)(t - 1) * HID) : y1;
        lstm_step<<<dim3(HID / 16), 32, 0, stream>>>(
            hp, t > 0, hstride,
            cbuf + (t & 1) * (BB * HID), cbuf + ((t + 1) & 1) * (BB * HID),
            Xk, Wr1t, y1 + (long)t * HID, inputs, t);
    }

    // --- Stage 6: projection  P = y1 @ Wp + bp  [2048,512] bf16 ---
    gemm_bt<1><<<dim3(EMBD / 64, MROWS / 128), 128, 0, stream>>>(
        y1, HID, Wpt, HID, MROWS, EMBD, HID, bp, nullptr, P, nullptr);

    // --- Stage 7: tied-embedding logits  out = P @ emb^T  [2048,32000] f32,
    //     with PAD rows replaced by onehot0 ---
    gemm_bt<2><<<dim3(VOCAB / 64, MROWS / 128), 128, 0, stream>>>(
        P, EMBD, embbf, EMBD, MROWS, VOCAB, EMBD, nullptr, out, nullptr, inputs);
}
